// RNN_86260123173877
// MI455X (gfx1250) — matrix operand order repaired and checked by reading, not compiled, not run on hardware
//
#include <hip/hip_runtime.h>
#include <math.h>
#include <stdint.h>

// ---------------------------------------------------------------------------
// Problem constants (match reference)
// ---------------------------------------------------------------------------
#define TT 64
#define BB 128
#define EE 1024
#define HH 1024
#define VV 10000
#define NEDGE 11          // NB-1 chain edges
#define MAX_NORM 25.0f

// WMMA types (CDNA5 gfx1250, wave32)
typedef __attribute__((ext_vector_type(16))) __bf16 v16bf;
typedef __attribute__((ext_vector_type(8)))  float  v8f;

// ---------------------------------------------------------------------------
// Helpers
// ---------------------------------------------------------------------------
__device__ __forceinline__ unsigned short f2bf(float f) {
    // round-to-nearest-even fp32 -> bf16 (bit-level, no __bf16 scalar math)
    unsigned int u = __float_as_uint(f);
    unsigned int r = u + 0x7FFFu + ((u >> 16) & 1u);
    return (unsigned short)(r >> 16);
}

__device__ __forceinline__ float sigmoidf_(float x) {
    return 1.0f / (1.0f + __expf(-x));
}

// A-fragment: 16x32 bf16 tile of row-major A (lda = K stride), contiguous k order (same as load_B_frag):
//   lane 0-15 : M = lane,    K base offset 0
//   lane16-31 : M = lane-16, K base offset 16
//   dword j (halves 2j,2j+1): K = 2j (+ base offset)
// 16 consecutive elements of the row per lane.
__device__ __forceinline__ v16bf load_A_frag(const unsigned short* A, int lda,
                                             int m0, int k0, int lane) {
    const unsigned short* row =
        A + (size_t)(m0 + (lane & 15)) * lda + k0 + ((lane >= 16) ? 16 : 0);
    v16bf a;
    uint32_t* ap = (uint32_t*)&a;
#pragma unroll
    for (int j = 0; j < 8; ++j) {
        int kk = 2 * j;
        ap[j] = *(const uint32_t*)(row + kk);
    }
    return a;
}

// B-fragment for D = A * W^T : B[k][n] = W[n][k], W row-major (ldw = K stride)
//   lane 0-15 : N = lane,    K = k0 + i      (i = half index 0..15)
//   lane16-31 : N = lane-16, K = k0 + 16 + i
// -> 16 consecutive bf16 per lane: two b128 loads.
__device__ __forceinline__ v16bf load_B_frag(const unsigned short* W, int ldw,
                                             int n0, int k0, int lane) {
    const unsigned short* p =
        W + (size_t)(n0 + (lane & 15)) * ldw + k0 + ((lane >= 16) ? 16 : 0);
    v16bf b;
    ((uint4*)&b)[0] = *(const uint4*)(p);
    ((uint4*)&b)[1] = *(const uint4*)(p + 8);
    return b;
}

#define WMMA_BF16(A, B, C) \
    __builtin_amdgcn_wmma_f32_16x16x32_bf16(false, (A), false, (B), (short)0, (C), false, false)

// ---------------------------------------------------------------------------
// f32 -> bf16 bulk convert
// ---------------------------------------------------------------------------
__global__ void cvt_bf16_kernel(const float* __restrict__ src,
                                unsigned short* __restrict__ dst, size_t n) {
    size_t i = (size_t)blockIdx.x * blockDim.x + threadIdx.x;
    if (i < n) dst[i] = f2bf(src[i]);
}

// ---------------------------------------------------------------------------
// Embedding gather -> bf16   x[t,b,e] = bf16(emb_W[tokens[t,b], e])
// ---------------------------------------------------------------------------
__global__ void embed_kernel(const int* __restrict__ tokens,
                             const float* __restrict__ embW,
                             unsigned short* __restrict__ x_bf) {
    size_t i = (size_t)blockIdx.x * blockDim.x + threadIdx.x;
    if (i >= (size_t)TT * BB * EE) return;
    int e  = (int)(i & (EE - 1));
    int tb = (int)(i >> 10);            // E = 1024
    int tok = tokens[tb];
    x_bf[i] = f2bf(embW[(size_t)tok * EE + e]);
}

// ---------------------------------------------------------------------------
// Zero-init hidden state (f32 + bf16)
// ---------------------------------------------------------------------------
__global__ void zero_h_kernel(float* __restrict__ hf, unsigned short* __restrict__ hb) {
    int i = blockIdx.x * blockDim.x + threadIdx.x;
    if (i < BB * HH) { hf[i] = 0.0f; hb[i] = 0; }
}

// ---------------------------------------------------------------------------
// Node 0:  c0 = sigmoid(x@Wxc^T + bc + h@Whc^T)
//          h' = c0*tanh(x@Wxh^T + bh + h@Whh^T) + (1-c0)*h
// Block 256 thr = 8 waves, tile 32(M) x 64(N).  Grid: (HH/64, BB/32)
// Double-buffered K loop: loads for k0+32 are issued before the WMMAs of k0.
// ---------------------------------------------------------------------------
__global__ __launch_bounds__(256) void node0_kernel(
    const unsigned short* __restrict__ x_bf,     // B x E (bf16)
    const unsigned short* __restrict__ hprev_bf, // B x H (bf16)
    const float* __restrict__ hprev_f,           // B x H (f32)
    const unsigned short* __restrict__ Wxc, const unsigned short* __restrict__ Whc,
    const unsigned short* __restrict__ Wxh, const unsigned short* __restrict__ Whh,
    const float* __restrict__ bc, const float* __restrict__ bh,
    float* __restrict__ ho_f, unsigned short* __restrict__ ho_b) {
    const int lane = threadIdx.x & 31, wave = threadIdx.x >> 5;
    const int m0 = blockIdx.y * 32 + (wave >> 2) * 16;
    const int n0 = blockIdx.x * 64 + (wave & 3) * 16;

    v8f accC = {}, accH = {};

    // prologue: fragments for k0 = 0
    v16bf ax  = load_A_frag(x_bf,     EE, m0, 0, lane);
    v16bf ah  = load_A_frag(hprev_bf, HH, m0, 0, lane);
    v16bf bxc = load_B_frag(Wxc, EE, n0, 0, lane);
    v16bf bxh = load_B_frag(Wxh, EE, n0, 0, lane);
    v16bf bhc = load_B_frag(Whc, HH, n0, 0, lane);
    v16bf bhh = load_B_frag(Whh, HH, n0, 0, lane);

    for (int k0 = 32; k0 < HH; k0 += 32) {   // E == H == 1024
        // issue next-slice loads first (no dependence on current WMMAs)
        v16bf ax2  = load_A_frag(x_bf,     EE, m0, k0, lane);
        v16bf ah2  = load_A_frag(hprev_bf, HH, m0, k0, lane);
        v16bf bxc2 = load_B_frag(Wxc, EE, n0, k0, lane);
        v16bf bxh2 = load_B_frag(Wxh, EE, n0, k0, lane);
        v16bf bhc2 = load_B_frag(Whc, HH, n0, k0, lane);
        v16bf bhh2 = load_B_frag(Whh, HH, n0, k0, lane);

        accC = WMMA_BF16(ax, bxc, accC);
        accH = WMMA_BF16(ax, bxh, accH);
        accC = WMMA_BF16(ah, bhc, accC);
        accH = WMMA_BF16(ah, bhh, accH);

        ax = ax2; ah = ah2; bxc = bxc2; bxh = bxh2; bhc = bhc2; bhh = bhh2;
    }
    accC = WMMA_BF16(ax, bxc, accC);
    accH = WMMA_BF16(ax, bxh, accH);
    accC = WMMA_BF16(ah, bhc, accC);
    accH = WMMA_BF16(ah, bhh, accH);

    const int n  = n0 + (lane & 15);
    const int mb = m0 + ((lane >> 4) << 3);
    const float bcv = bc[n], bhv = bh[n];
#pragma unroll
    for (int r = 0; r < 8; ++r) {
        int   m  = mb + r;
        float hp = hprev_f[m * HH + n];
        float c0 = sigmoidf_(accC[r] + bcv);
        float hv = c0 * tanhf(accH[r] + bhv) + (1.0f - c0) * hp;   // F0 = tanh
        ho_f[m * HH + n] = hv;
        ho_b[m * HH + n] = f2bf(hv);
    }
}

// ---------------------------------------------------------------------------
// Chain edge:  c = sigmoid(h@Wc^T);  h' = c*act(h@Wh^T) + (1-c)*h
// act: 0 = tanh, 1 = relu.  out_bf != nullptr on the last edge (outputs slice).
// Double-buffered K loop.
// ---------------------------------------------------------------------------
__global__ __launch_bounds__(256) void edge_kernel(
    const unsigned short* __restrict__ h_bf, const float* __restrict__ h_f,
    const unsigned short* __restrict__ Wc,   const unsigned short* __restrict__ Wh,
    float* __restrict__ ho_f, unsigned short* __restrict__ ho_b,
    unsigned short* __restrict__ out_bf, int act_relu) {
    const int lane = threadIdx.x & 31, wave = threadIdx.x >> 5;
    const int m0 = blockIdx.y * 32 + (wave >> 2) * 16;
    const int n0 = blockIdx.x * 64 + (wave & 3) * 16;

    v8f accC = {}, accH = {};

    v16bf a  = load_A_frag(h_bf, HH, m0, 0, lane);
    v16bf wc = load_B_frag(Wc, HH, n0, 0, lane);
    v16bf wh = load_B_frag(Wh, HH, n0, 0, lane);

    for (int k0 = 32; k0 < HH; k0 += 32) {
        v16bf a2  = load_A_frag(h_bf, HH, m0, k0, lane);
        v16bf wc2 = load_B_frag(Wc, HH, n0, k0, lane);
        v16bf wh2 = load_B_frag(Wh, HH, n0, k0, lane);

        accC = WMMA_BF16(a, wc, accC);
        accH = WMMA_BF16(a, wh, accH);

        a = a2; wc = wc2; wh = wh2;
    }
    accC = WMMA_BF16(a, wc, accC);
    accH = WMMA_BF16(a, wh, accH);

    const int n  = n0 + (lane & 15);
    const int mb = m0 + ((lane >> 4) << 3);
#pragma unroll
    for (int r = 0; r < 8; ++r) {
        int   m  = mb + r;
        float hi = h_f[m * HH + n];
        float c  = sigmoidf_(accC[r]);
        float av = act_relu ? fmaxf(accH[r], 0.0f) : tanhf(accH[r]);
        float ho = c * av + (1.0f - c) * hi;
        ho_f[m * HH + n] = ho;
        unsigned short hb = f2bf(ho);
        ho_b[m * HH + n] = hb;
        if (out_bf) out_bf[m * HH + n] = hb;   // outputs[t] (pre-clip) for decoder
    }
}

// ---------------------------------------------------------------------------
// Per-row norm clip: h_state = h * min(1, MAX_NORM/||h||).  One block per row.
// ---------------------------------------------------------------------------
__global__ __launch_bounds__(256) void clip_kernel(
    const float* __restrict__ hin,
    float* __restrict__ hstate_f, unsigned short* __restrict__ hstate_b) {
    __shared__ float red[256];
    const int row = blockIdx.x;
    float s = 0.0f;
    for (int i = threadIdx.x; i < HH; i += 256) {
        float v = hin[row * HH + i];
        s += v * v;
    }
    red[threadIdx.x] = s;
    __syncthreads();
    for (int off = 128; off > 0; off >>= 1) {
        if (threadIdx.x < off) red[threadIdx.x] += red[threadIdx.x + off];
        __syncthreads();
    }
    float norm  = sqrtf(red[0]);
    float scale = (norm > MAX_NORM) ? (MAX_NORM / norm) : 1.0f;
    for (int i = threadIdx.x; i < HH; i += 256) {
        float v = hin[row * HH + i] * scale;
        hstate_f[row * HH + i] = v;
        hstate_b[row * HH + i] = f2bf(v);
    }
}

// ---------------------------------------------------------------------------
// Decoder: out[m, v] = outputs_bf[m,:] . dec_bf[v,:] + dec_b[v]
// M = T*B = 8192, N = V = 10000 (multiple of 16).  Grid: (ceil(V/64), M/32)
// Double-buffered K loop.
// ---------------------------------------------------------------------------
__global__ __launch_bounds__(256) void decoder_kernel(
    const unsigned short* __restrict__ A,   // (T*B) x H bf16
    const unsigned short* __restrict__ Wd,  // V x H bf16
    const float* __restrict__ bias, float* __restrict__ out) {
    const int lane = threadIdx.x & 31, wave = threadIdx.x >> 5;
    const int n0 = blockIdx.x * 64 + (wave & 3) * 16;
    if (n0 >= VV) return;                    // wave-uniform guard (V % 16 == 0)
    const int m0 = blockIdx.y * 32 + (wave >> 2) * 16;

    v8f acc = {};
    v16bf a = load_A_frag(A,  HH, m0, 0, lane);
    v16bf b = load_B_frag(Wd, HH, n0, 0, lane);
    for (int k0 = 32; k0 < HH; k0 += 32) {
        v16bf a2 = load_A_frag(A,  HH, m0, k0, lane);
        v16bf b2 = load_B_frag(Wd, HH, n0, k0, lane);
        acc = WMMA_BF16(a, b, acc);
        a = a2; b = b2;
    }
    acc = WMMA_BF16(a, b, acc);

    const int n  = n0 + (lane & 15);
    const int mb = m0 + ((lane >> 4) << 3);
    const float bv = bias[n];
#pragma unroll
    for (int r = 0; r < 8; ++r) {
        size_t m = (size_t)(mb + r);
        out[m * VV + n] = acc[r] + bv;
    }
}

// ---------------------------------------------------------------------------
// Copy final hidden state into tail of d_out
// ---------------------------------------------------------------------------
__global__ void copy_hT_kernel(const float* __restrict__ hf, float* __restrict__ dst) {
    int i = blockIdx.x * blockDim.x + threadIdx.x;
    if (i < BB * HH) dst[i] = hf[i];
}

// ---------------------------------------------------------------------------
// Launch
// ---------------------------------------------------------------------------
extern "C" void kernel_launch(void* const* d_in, const int* in_sizes, int n_in,
                              void* d_out, int out_size, void* d_ws, size_t ws_size,
                              hipStream_t stream) {
    (void)in_sizes; (void)n_in; (void)out_size; (void)ws_size;

    const int*   tokens = (const int*)  d_in[0];
    const float* emb_W  = (const float*)d_in[1];
    const float* wxc_W  = (const float*)d_in[2];
    const float* wxc_b  = (const float*)d_in[3];
    const float* wxh_W  = (const float*)d_in[4];
    const float* wxh_b  = (const float*)d_in[5];
    const float* w_hc   = (const float*)d_in[6];
    const float* w_hh   = (const float*)d_in[7];
    const float* edge_h = (const float*)d_in[8];
    const float* edge_c = (const float*)d_in[9];
    const float* dec_W  = (const float*)d_in[10];
    const float* dec_b  = (const float*)d_in[11];
    float* out = (float*)d_out;

    // -------- workspace carve-up (~106 MB total) --------
    uint8_t* base = (uint8_t*)d_ws;
    size_t off = 0;
    auto alloc = [&](size_t bytes) -> void* {
        void* p = base + off;
        off += (bytes + 255) & ~(size_t)255;
        return p;
    };
    unsigned short* x_bf    = (unsigned short*)alloc((size_t)TT * BB * EE * 2);
    unsigned short* wxc_bf  = (unsigned short*)alloc((size_t)HH * EE * 2);
    unsigned short* wxh_bf  = (unsigned short*)alloc((size_t)HH * EE * 2);
    unsigned short* whc_bf  = (unsigned short*)alloc((size_t)HH * HH * 2);
    unsigned short* whh_bf  = (unsigned short*)alloc((size_t)HH * HH * 2);
    unsigned short* eh_bf   = (unsigned short*)alloc((size_t)NEDGE * HH * HH * 2);
    unsigned short* ec_bf   = (unsigned short*)alloc((size_t)NEDGE * HH * HH * 2);
    unsigned short* dec_bf  = (unsigned short*)alloc((size_t)VV * HH * 2);
    unsigned short* outs_bf = (unsigned short*)alloc((size_t)TT * BB * HH * 2);
    float*          hS_f    = (float*)         alloc((size_t)BB * HH * 4);
    unsigned short* hS_b    = (unsigned short*)alloc((size_t)BB * HH * 2);
    float*          h0_f    = (float*)         alloc((size_t)BB * HH * 4);
    unsigned short* h0_b    = (unsigned short*)alloc((size_t)BB * HH * 2);
    float*          h1_f    = (float*)         alloc((size_t)BB * HH * 4);
    unsigned short* h1_b    = (unsigned short*)alloc((size_t)BB * HH * 2);

    // -------- one-shot conversions (served from L2 thereafter) --------
    auto cvt = [&](const float* s, unsigned short* d, size_t n) {
        cvt_bf16_kernel<<<(unsigned)((n + 255) / 256), 256, 0, stream>>>(s, d, n);
    };
    cvt(wxc_W, wxc_bf, (size_t)HH * EE);
    cvt(wxh_W, wxh_bf, (size_t)HH * EE);
    cvt(w_hc,  whc_bf, (size_t)HH * HH);
    cvt(w_hh,  whh_bf, (size_t)HH * HH);
    cvt(edge_h, eh_bf, (size_t)NEDGE * HH * HH);
    cvt(edge_c, ec_bf, (size_t)NEDGE * HH * HH);
    cvt(dec_W, dec_bf, (size_t)VV * HH);

    embed_kernel<<<(unsigned)(((size_t)TT * BB * EE + 255) / 256), 256, 0, stream>>>(
        tokens, emb_W, x_bf);

    zero_h_kernel<<<(BB * HH + 255) / 256, 256, 0, stream>>>(hS_f, hS_b);

    // -------- recurrence: 64 sequential steps --------
    const dim3 gemm_grid(HH / 64, BB / 32);   // 16 x 4 blocks, 8 waves each
    for (int t = 0; t < TT; ++t) {
        const unsigned short* xt = x_bf + (size_t)t * BB * EE;

        node0_kernel<<<gemm_grid, 256, 0, stream>>>(
            xt, hS_b, hS_f, wxc_bf, whc_bf, wxh_bf, whh_bf,
            wxc_b, wxh_b, h0_f, h0_b);

        // chain edges 0..10, ping-pong h0 <-> h1
        for (int j = 0; j < NEDGE; ++j) {
            const float*          in_f = (j & 1) ? h1_f : h0_f;
            const unsigned short* in_b = (j & 1) ? h1_b : h0_b;
            float*                ot_f = (j & 1) ? h0_f : h1_f;
            unsigned short*       ot_b = (j & 1) ? h0_b : h1_b;
            unsigned short* slice =
                (j == NEDGE - 1) ? (outs_bf + (size_t)t * BB * HH) : nullptr;
            int act_relu = (j % 2 == 0) ? 0 : 1;   // EDGE_ACTS: even=tanh, odd=relu
            edge_kernel<<<gemm_grid, 256, 0, stream>>>(
                in_b, in_f, ec_bf + (size_t)j * HH * HH, eh_bf + (size_t)j * HH * HH,
                ot_f, ot_b, slice, act_relu);
        }

        // final edge output lives in h1 (NEDGE odd); clip into carried state
        clip_kernel<<<BB, 256, 0, stream>>>(h1_f, hS_f, hS_b);
    }

    // -------- decoder GEMM: (T*B=8192) x H x V --------
    decoder_kernel<<<dim3((VV + 63) / 64, (TT * BB) / 32), 256, 0, stream>>>(
        outs_bf, dec_bf, dec_b, out);

    // -------- hT tail --------
    copy_hT_kernel<<<(BB * HH + 255) / 256, 256, 0, stream>>>(
        hS_f, out + (size_t)TT * BB * VV);
}
